// TeacherModel_79259326480957
// MI455X (gfx1250) — matrix operand order repaired and checked by reading, not compiled, not run on hardware
//
#include <hip/hip_runtime.h>

#define Bq 2
#define Nn 96
#define Hh 768
#define HG 384           // H2
#define Pp 128
#define NHEADS 12
#define Dd 64
#define Tt 17
#define NN2 (Nn * Nn)    // 9216
#define ROWS (Bq * NN2)  // 18432
#define BN (Bq * Nn)     // 192
#define G3 (3 * HG)      // 1152
#define CIN_L (2 * Hh + Pp)  // 1664
#define CIN_P (2 * Hh)       // 1536
#define CIN_F (4 * Hh + Pp)  // 3200

typedef __attribute__((ext_vector_type(16))) __bf16 bf16x16;
typedef __attribute__((ext_vector_type(8)))  __bf16 bf16x8;
typedef __attribute__((ext_vector_type(8)))  float  f32x8;

// ---------------- device helpers ----------------

__device__ __forceinline__ __bf16 f2bf(float f) {
    unsigned u = __float_as_uint(f);
    u = u + 0x7FFFu + ((u >> 16) & 1u);                 // round-to-nearest-even
    unsigned short s = (unsigned short)(u >> 16);
    return __builtin_bit_cast(__bf16, s);
}

// A fragment (16x32 bf16, M = lane%16): lanes<16 hold K {k0..k0+15},
// lanes>=16 hold K {k0+16..k0+31} -> two contiguous 16B loads.
__device__ __forceinline__ bf16x16 load_a_frag(const __bf16* __restrict__ rowptr,
                                               int k0, int hi16) {
    const bf16x8 lo = *reinterpret_cast<const bf16x8*>(rowptr + k0 + (hi16 ? 16 : 0));
    const bf16x8 hi = *reinterpret_cast<const bf16x8*>(rowptr + k0 + 8 + (hi16 ? 16 : 0));
    bf16x16 r;
#pragma unroll
    for (int i = 0; i < 8; ++i) { r[i] = lo[i]; r[i + 8] = hi[i]; }
    return r;
}

// B fragment (32x16 bf16). Weight stored N-major (out x in): lane owns column
// N = n0 + lane%16, K = k0 + (lane>=16 ? 16:0) + e -> 16 contiguous bf16.
__device__ __forceinline__ bf16x16 load_b_frag(const __bf16* __restrict__ W, int ldw,
                                               int n0, int k0, int lane) {
    const __bf16* rp = W + (size_t)(n0 + (lane & 15)) * ldw + k0 + ((lane >= 16) ? 16 : 0);
    const bf16x8 lo = *reinterpret_cast<const bf16x8*>(rp);
    const bf16x8 hi = *reinterpret_cast<const bf16x8*>(rp + 8);
    bf16x16 r;
#pragma unroll
    for (int i = 0; i < 8; ++i) { r[i] = lo[i]; r[i + 8] = hi[i]; }
    return r;
}

__device__ __forceinline__ f32x8 wmma_bf(bf16x16 a, bf16x16 b, f32x8 c) {
    return __builtin_amdgcn_wmma_f32_16x16x32_bf16(false, a, false, b, (short)0, c,
                                                   false, false);
}

__device__ __forceinline__ float sigmoidf_(float x) { return 1.0f / (1.0f + expf(-x)); }

// ---------------- generic GEMM: C[M,N] = (A_bf16[M,K] @ W_bf16[N,K]^T + bias) * alpha
// One wave computes a 16x64 output strip (4 accumulators, A-fragment reused 4x).
// All fragment loads are issued before the 4 WMMAs so the scheduler can use
// partial s_wait_loadcnt and overlap WMMA with outstanding loads.
// grid = (M/16, N/64), block = 32
__global__ void __launch_bounds__(32) gemm_bf16_kernel(
    const __bf16* __restrict__ A, int lda,
    const __bf16* __restrict__ W, int ldw,
    const float* __restrict__ bias,
    float* __restrict__ C, int ldc,
    int K, float alpha) {
    const int m0 = blockIdx.x * 16, n0 = blockIdx.y * 64;
    const int lane = threadIdx.x;
    const int hi16 = (lane >= 16) ? 1 : 0;
    const __bf16* arow = A + (size_t)(m0 + (lane & 15)) * lda;
    f32x8 acc[4];
#pragma unroll
    for (int t = 0; t < 4; ++t) acc[t] = (f32x8){0, 0, 0, 0, 0, 0, 0, 0};
    for (int k0 = 0; k0 < K; k0 += 32) {
        bf16x16 a  = load_a_frag(arow, k0, hi16);
        bf16x16 b0 = load_b_frag(W, ldw, n0 +  0, k0, lane);
        bf16x16 b1 = load_b_frag(W, ldw, n0 + 16, k0, lane);
        bf16x16 b2 = load_b_frag(W, ldw, n0 + 32, k0, lane);
        bf16x16 b3 = load_b_frag(W, ldw, n0 + 48, k0, lane);
        acc[0] = wmma_bf(a, b0, acc[0]);
        acc[1] = wmma_bf(a, b1, acc[1]);
        acc[2] = wmma_bf(a, b2, acc[2]);
        acc[3] = wmma_bf(a, b3, acc[3]);
    }
#pragma unroll
    for (int t = 0; t < 4; ++t) {
        const int n = n0 + 16 * t + (lane & 15);
        const float bvv = bias ? bias[n] : 0.0f;
#pragma unroll
        for (int r = 0; r < 8; ++r) {
            int m = m0 + r + (hi16 ? 8 : 0);
            C[(size_t)m * ldc + n] = (acc[t][r] + bvv) * alpha;
        }
    }
}

// ---------------- elementwise / small kernels ----------------

__global__ void cvt_kernel(const float* __restrict__ s, __bf16* __restrict__ d, int n) {
    int i = blockIdx.x * 256 + threadIdx.x;
    if (i < n) d[i] = f2bf(s[i]);
}

__global__ void din_kernel(const float* __restrict__ sent, const float* __restrict__ root,
                           float* __restrict__ din, __bf16* __restrict__ din_bf) {
    int idx = blockIdx.x * 256 + threadIdx.x;
    if (idx >= BN * Hh) return;
    int h = idx % Hh;
    int t = (idx / Hh) % Nn;
    int b = idx / (Hh * Nn);
    float v = (t == 0) ? root[h] : sent[((size_t)b * (Nn - 1) + (t - 1)) * Hh + h];
    din[idx] = v;
    din_bf[idx] = f2bf(v);
}

__global__ void nodes_init_kernel(const float* __restrict__ dout,
                                  float* __restrict__ nodes, __bf16* __restrict__ nodes_bf) {
    int idx = blockIdx.x * 256 + threadIdx.x;
    if (idx >= BN * Hh) return;
    float v = dout[idx];
    nodes[idx] = v;
    nodes_bf[idx] = f2bf(v);
}

__global__ void multi_kernel(const float* __restrict__ din, const float* __restrict__ dout,
                             __bf16* __restrict__ multi_bf) {
    int idx = blockIdx.x * 256 + threadIdx.x;
    if (idx >= BN * CIN_P) return;
    int c = idx % CIN_P;
    int row = idx / CIN_P;
    float v = (c < Hh) ? din[(size_t)row * Hh + c] : dout[(size_t)row * Hh + (c - Hh)];
    multi_bf[idx] = f2bf(v);
}

// const_path + struct_path init. grid = ROWS, block = 128
__global__ void const_sp_kernel(const int* __restrict__ speakers, const int* __restrict__ turns,
                                const int* __restrict__ graphs,
                                const float* __restrict__ emb_sp, const float* __restrict__ emb_tn,
                                const float* __restrict__ emb_pos,
                                const float* __restrict__ normal_type,
                                float* __restrict__ cpath, float* __restrict__ sp,
                                __bf16* __restrict__ sp_bf) {
    int rho = blockIdx.x;
    int p = threadIdx.x;
    int b = rho / NN2, rem = rho % NN2, ii = rem / Nn, jj = rem % Nn;
    float cv;
    if (p < 32) cv = emb_sp[speakers[rho] * 32 + p];
    else if (p < 64) cv = emb_tn[turns[rho] * 32 + (p - 32)];
    else {
        int rel = jj - ii;
        if (rel > 11) rel = 11;
        if (rel < -11) rel = -11;
        rel += 11;
        cv = emb_pos[rel * 64 + (p - 64)];
    }
    size_t o = (size_t)rho * Pp + p;
    cpath[o] = cv;
    int g = graphs[rho] + graphs[b * NN2 + jj * Nn + ii];
    float s = (g == 0) ? 0.0f : normal_type[(size_t)(g - 1) * Pp + p];
    sp[o] = s;
    sp_bf[o] = f2bf(s);
}

// symmetrize sp (lower -> upper, in place, race-free), refresh sp_bf,
// paths = LN(sp + const_path). grid = ROWS, block = 128
__global__ void paths_ln_kernel(float* __restrict__ sp, __bf16* __restrict__ sp_bf,
                                const float* __restrict__ cpath,
                                const float* __restrict__ pln_s, const float* __restrict__ pln_b,
                                __bf16* __restrict__ paths_bf) {
    __shared__ float red[128];
    int rho = blockIdx.x;
    int p = threadIdx.x;
    int b = rho / NN2, rem = rho % NN2, ii = rem / Nn, jj = rem % Nn;
    size_t o = (size_t)rho * Pp + p;
    float v;
    if (ii < jj) {
        v = sp[((size_t)(b * NN2 + jj * Nn + ii)) * Pp + p];  // read lower (never written here)
        sp[o] = v;
    } else {
        v = sp[o];
    }
    sp_bf[o] = f2bf(v);
    float x = v + cpath[o];
    red[p] = x;
    __syncthreads();
    for (int st = 64; st > 0; st >>= 1) { if (p < st) red[p] += red[p + st]; __syncthreads(); }
    float mean = red[0] * (1.0f / 128.0f);
    __syncthreads();
    float dlt = x - mean;
    red[p] = dlt * dlt;
    __syncthreads();
    for (int st = 64; st > 0; st >>= 1) { if (p < st) red[p] += red[p + st]; __syncthreads(); }
    float var = red[0] * (1.0f / 128.0f);
    float y = dlt * rsqrtf(var + 1e-5f) * pln_s[p] + pln_b[p];
    paths_bf[o] = f2bf(y);
}

// Bidirectional GRU recurrence. grid = 4 (b*2+dir), block = 384, h in LDS.
__global__ void gru_scan_kernel(const float* __restrict__ xw_f, const float* __restrict__ xw_b,
                                const float* __restrict__ whh_f, const float* __restrict__ whh_b,
                                const float* __restrict__ bhh_f, const float* __restrict__ bhh_b,
                                float* __restrict__ dout) {
    int b = blockIdx.x >> 1;
    int rev = blockIdx.x & 1;
    const float* xw = rev ? xw_b : xw_f;
    const float* whh = rev ? whh_b : whh_f;
    const float* bhh = rev ? bhh_b : bhh_f;
    __shared__ float h[HG];
    int i = threadIdx.x;
    h[i] = 0.0f;
    __syncthreads();
    const float* wr_ = whh + (size_t)i * HG;
    const float* wz_ = whh + (size_t)(HG + i) * HG;
    const float* wn_ = whh + (size_t)(2 * HG + i) * HG;
    for (int step = 0; step < Nn; ++step) {
        int t = rev ? (Nn - 1 - step) : step;
        const float* xwt = xw + ((size_t)(b * Nn + t)) * G3;
        float gr = bhh[i], gz = bhh[HG + i], gn = bhh[2 * HG + i];
        for (int c = 0; c < HG; ++c) {
            float hv = h[c];
            gr += wr_[c] * hv;
            gz += wz_[c] * hv;
            gn += wn_[c] * hv;
        }
        float r = sigmoidf_(xwt[i] + gr);
        float z = sigmoidf_(xwt[HG + i] + gz);
        float nn = tanhf(xwt[2 * HG + i] + r * gn);
        float hn = (1.0f - z) * nn + z * h[i];
        __syncthreads();
        h[i] = hn;
        dout[((size_t)(b * Nn + t)) * Hh + rev * HG + i] = hn;
        __syncthreads();
    }
}

// Structure-aware attention. grid = B*HEADS*N, block = 128
__global__ void attn_kernel(const float* __restrict__ q, const float* __restrict__ k,
                            const float* __restrict__ v, const float* __restrict__ sk,
                            const float* __restrict__ sv, const int* __restrict__ edu_nums,
                            float* __restrict__ actx) {
    __shared__ float qv[Dd];
    __shared__ float sc[128];
    __shared__ float red[128];
    int bid = blockIdx.x;
    int n = bid % Nn;
    int hh = (bid / Nn) % NHEADS;
    int b = bid / (Nn * NHEADS);
    int tid = threadIdx.x;
    if (tid < Dd) qv[tid] = q[((size_t)(b * Nn + n)) * Hh + hh * Dd + tid];
    __syncthreads();
    int edn = edu_nums[b];
    float s = -3.0e38f;
    if (tid < Nn) {
        const float* kp = k + ((size_t)(b * Nn + tid)) * Hh + hh * Dd;
        const float* skp = sk + ((size_t)((b * NN2) + n * Nn + tid)) * Dd;
        float acc = 0.0f;
        for (int d = 0; d < Dd; ++d) acc += qv[d] * (kp[d] + skp[d]);
        s = acc + ((tid < edn) ? 0.0f : -1.0e9f);
    }
    sc[tid] = s;
    red[tid] = s;
    __syncthreads();
    for (int st = 64; st > 0; st >>= 1) { if (tid < st) red[tid] = fmaxf(red[tid], red[tid + st]); __syncthreads(); }
    float mx = red[0];
    __syncthreads();
    float e = (tid < Nn) ? expf(sc[tid] - mx) : 0.0f;
    sc[tid] = e;
    red[tid] = e;
    __syncthreads();
    for (int st = 64; st > 0; st >>= 1) { if (tid < st) red[tid] += red[tid + st]; __syncthreads(); }
    float inv = 1.0f / red[0];
    if (tid < Dd) {
        const float* vb = v + (size_t)(b * Nn) * Hh + hh * Dd + tid;
        const float* svb = sv + ((size_t)(b * NN2) + n * Nn) * Dd + tid;
        float o = 0.0f;
        for (int m = 0; m < Nn; ++m) o += sc[m] * (vb[(size_t)m * Hh] + svb[(size_t)m * Dd]);
        actx[((size_t)(b * Nn + n)) * Hh + hh * Dd + tid] = o * inv;
    }
}

// nodes = LN(nodes + relu(proj)). grid = 192, block = 256
__global__ void resid_ln_kernel(float* __restrict__ nodes, __bf16* __restrict__ nodes_bf,
                                const float* __restrict__ proj,
                                const float* __restrict__ ln_s, const float* __restrict__ ln_b) {
    __shared__ float red[256];
    __shared__ float xbuf[Hh];
    int row = blockIdx.x;
    int tid = threadIdx.x;
    for (int c = tid; c < Hh; c += 256) {
        float x = nodes[(size_t)row * Hh + c] + fmaxf(proj[(size_t)row * Hh + c], 0.0f);
        xbuf[c] = x;
    }
    __syncthreads();
    float ls = 0.0f;
    for (int c = tid; c < Hh; c += 256) ls += xbuf[c];
    red[tid] = ls;
    __syncthreads();
    for (int st = 128; st > 0; st >>= 1) { if (tid < st) red[tid] += red[tid + st]; __syncthreads(); }
    float mean = red[0] * (1.0f / (float)Hh);
    __syncthreads();
    float lv = 0.0f;
    for (int c = tid; c < Hh; c += 256) { float d = xbuf[c] - mean; lv += d * d; }
    red[tid] = lv;
    __syncthreads();
    for (int st = 128; st > 0; st >>= 1) { if (tid < st) red[tid] += red[tid + st]; __syncthreads(); }
    float inv = rsqrtf(red[0] * (1.0f / (float)Hh) + 1e-5f);
    for (int c = tid; c < Hh; c += 256) {
        float y = (xbuf[c] - mean) * inv * ln_s[c] + ln_b[c];
        nodes[(size_t)row * Hh + c] = y;
        nodes_bf[(size_t)row * Hh + c] = f2bf(y);
    }
}

// Fused path-GRU update: r/z/c-pre/u-pre via WMMA into LDS, gated update on tril.
// 8 waves <-> exactly 8 (plane, 64-wide half) strip tasks; A fragment reused 4x.
// grid = ROWS/16, block = 256
__global__ void __launch_bounds__(256) path_update_kernel(
    const __bf16* __restrict__ nodes_bf, const __bf16* __restrict__ sp_bf,
    const __bf16* __restrict__ wr_bf, const __bf16* __restrict__ wz_bf,
    const __bf16* __restrict__ wc_bf, const __bf16* __restrict__ wu_bf,
    const float* __restrict__ br, const float* __restrict__ bz,
    const float* __restrict__ bc, const float* __restrict__ bu,
    float* __restrict__ sp) {
    __shared__ float pl[4][16][Pp];  // 32 KB
    const int m0 = blockIdx.x * 16;
    const int tid = threadIdx.x;
    const int lane = tid & 31, wave = tid >> 5;
    const int hi16 = (lane >= 16) ? 1 : 0;
    const int rho = m0 + (lane & 15);
    const int b = rho / NN2, rem = rho % NN2, ii = rem / Nn, jj = rem % Nn;
    const __bf16* nrow_i = nodes_bf + (size_t)(b * Nn + ii) * Hh;
    const __bf16* nrow_j = nodes_bf + (size_t)(b * Nn + jj) * Hh;  // pair = [nodes_j | nodes_i]
    const __bf16* srow = sp_bf + (size_t)rho * Pp;

    const int qn = wave >> 1;          // plane: 0=r, 1=z, 2=c-pre, 3=u-pre
    const int n0 = (wave & 1) * 64;    // 64-wide half of the 128 outputs
    const __bf16* W;
    const float* bias;
    int Kq;
    if (qn == 0) { W = wr_bf; bias = br; Kq = CIN_L; }
    else if (qn == 1) { W = wz_bf; bias = bz; Kq = CIN_L; }
    else if (qn == 2) { W = wc_bf; bias = bc; Kq = CIN_P; }
    else { W = wu_bf; bias = bu; Kq = Pp; }
    f32x8 acc[4];
#pragma unroll
    for (int t = 0; t < 4; ++t) acc[t] = (f32x8){0, 0, 0, 0, 0, 0, 0, 0};
    for (int k0 = 0; k0 < Kq; k0 += 32) {
        const __bf16* arow;
        int kk;
        if (qn == 3) { arow = srow; kk = k0; }
        else if (k0 < Hh) { arow = nrow_j; kk = k0; }
        else if (k0 < 2 * Hh) { arow = nrow_i; kk = k0 - Hh; }
        else { arow = srow; kk = k0 - 2 * Hh; }
        bf16x16 a  = load_a_frag(arow, kk, hi16);
        bf16x16 b0 = load_b_frag(W, Kq, n0 +  0, k0, lane);
        bf16x16 b1 = load_b_frag(W, Kq, n0 + 16, k0, lane);
        bf16x16 b2 = load_b_frag(W, Kq, n0 + 32, k0, lane);
        bf16x16 b3 = load_b_frag(W, Kq, n0 + 48, k0, lane);
        acc[0] = wmma_bf(a, b0, acc[0]);
        acc[1] = wmma_bf(a, b1, acc[1]);
        acc[2] = wmma_bf(a, b2, acc[2]);
        acc[3] = wmma_bf(a, b3, acc[3]);
    }
#pragma unroll
    for (int t = 0; t < 4; ++t) {
        const int n = n0 + 16 * t + (lane & 15);
        const float bvv = bias[n];
#pragma unroll
        for (int r = 0; r < 8; ++r) pl[qn][r + (hi16 ? 8 : 0)][n] = acc[t][r] + bvv;
    }
    __syncthreads();
    for (int idx = tid; idx < 16 * Pp; idx += 256) {
        int row = idx / Pp, p = idx % Pp;
        int r2 = m0 + row;
        int rem2 = r2 % NN2, i2 = rem2 / Nn, j2 = rem2 % Nn;
        if (i2 <= j2) continue;  // tril only
        float rg = sigmoidf_(pl[0][row][p]);
        float zg = sigmoidf_(pl[1][row][p]);
        float ug = tanhf(pl[2][row][p] + rg * pl[3][row][p]);
        size_t o = (size_t)r2 * Pp + p;
        float old = sp[o];
        sp[o] = zg * old + (1.0f - zg) * ug;
    }
}

// Fused link/label head: hidden(16x768) via WMMA into LDS + tiny second GEMM.
// Each wave does 64-wide strips (A reuse 4x). grid = (ROWS/16, 2), block = 256.
__global__ void __launch_bounds__(256) head_kernel(
    const __bf16* __restrict__ multi_bf, const __bf16* __restrict__ sp_bf,
    const __bf16* __restrict__ w1_link, const __bf16* __restrict__ w1_lab,
    const float* __restrict__ b1_link, const float* __restrict__ b1_lab,
    const float* __restrict__ link_w2, const float* __restrict__ link_b2,
    const float* __restrict__ lab_w2, const float* __restrict__ lab_b2,
    float* __restrict__ link_out, float* __restrict__ label_out) {
    __shared__ float hid[16][Hh];  // 48 KB
    const int m0 = blockIdx.x * 16;
    const int head = blockIdx.y;
    const __bf16* W1 = head ? w1_lab : w1_link;
    const float* B1 = head ? b1_lab : b1_link;
    const int tid = threadIdx.x;
    const int lane = tid & 31, wave = tid >> 5;
    const int hi16 = (lane >= 16) ? 1 : 0;
    const int rho = m0 + (lane & 15);
    const int b = rho / NN2, rem = rho % NN2, ii = rem / Nn, jj = rem % Nn;
    const __bf16* mrow_i = multi_bf + (size_t)(b * Nn + ii) * CIN_P;
    const __bf16* mrow_j = multi_bf + (size_t)(b * Nn + jj) * CIN_P;  // feats = [multi_j|multi_i|sp]
    const __bf16* srow = sp_bf + (size_t)rho * Pp;
    for (int g = wave; g < Hh / 64; g += 8) {   // 12 strips of 64
        const int n0 = g * 64;
        f32x8 acc[4];
#pragma unroll
        for (int t = 0; t < 4; ++t) acc[t] = (f32x8){0, 0, 0, 0, 0, 0, 0, 0};
        for (int k0 = 0; k0 < CIN_F; k0 += 32) {
            const __bf16* arow;
            int kk;
            if (k0 < CIN_P) { arow = mrow_j; kk = k0; }
            else if (k0 < 2 * CIN_P) { arow = mrow_i; kk = k0 - CIN_P; }
            else { arow = srow; kk = k0 - 2 * CIN_P; }
            bf16x16 a  = load_a_frag(arow, kk, hi16);
            bf16x16 b0 = load_b_frag(W1, CIN_F, n0 +  0, k0, lane);
            bf16x16 b1 = load_b_frag(W1, CIN_F, n0 + 16, k0, lane);
            bf16x16 b2 = load_b_frag(W1, CIN_F, n0 + 32, k0, lane);
            bf16x16 b3 = load_b_frag(W1, CIN_F, n0 + 48, k0, lane);
            acc[0] = wmma_bf(a, b0, acc[0]);
            acc[1] = wmma_bf(a, b1, acc[1]);
            acc[2] = wmma_bf(a, b2, acc[2]);
            acc[3] = wmma_bf(a, b3, acc[3]);
        }
#pragma unroll
        for (int t = 0; t < 4; ++t) {
            const int n = n0 + 16 * t + (lane & 15);
            const float bvv = B1[n];
#pragma unroll
            for (int r = 0; r < 8; ++r) hid[r + (hi16 ? 8 : 0)][n] = tanhf(acc[t][r] + bvv);
        }
    }
    __syncthreads();
    if (head == 0) {
        if (tid < 16) {
            float s = link_b2[0];
            for (int c = 0; c < Hh; ++c) s += hid[tid][c] * link_w2[c];
            link_out[m0 + tid] = s;
        }
    } else {
        for (int idx = tid; idx < 16 * Tt; idx += 256) {
            int row = idx / Tt, t = idx % Tt;
            float s = lab_b2[t];
            for (int c = 0; c < Hh; ++c) s += hid[row][c] * lab_w2[(size_t)t * Hh + c];
            label_out[(size_t)(m0 + row) * Tt + t] = s;
        }
    }
}

// ---------------- host orchestration ----------------

extern "C" void kernel_launch(void* const* d_in, const int* in_sizes, int n_in,
                              void* d_out, int out_size, void* d_ws, size_t ws_size,
                              hipStream_t stream) {
    (void)in_sizes; (void)n_in; (void)out_size; (void)ws_size;
    const float* sentences   = (const float*)d_in[0];
    const int*   speakers    = (const int*)d_in[1];
    const int*   turns       = (const int*)d_in[2];
    const int*   graphs      = (const int*)d_in[3];
    const int*   edu_nums    = (const int*)d_in[4];
    const float* root        = (const float*)d_in[5];
    const float* gru_wih_f   = (const float*)d_in[6];
    const float* gru_whh_f   = (const float*)d_in[7];
    const float* gru_bih_f   = (const float*)d_in[8];
    const float* gru_bhh_f   = (const float*)d_in[9];
    const float* gru_wih_b   = (const float*)d_in[10];
    const float* gru_whh_b   = (const float*)d_in[11];
    const float* gru_bih_b   = (const float*)d_in[12];
    const float* gru_bhh_b   = (const float*)d_in[13];
    const float* emb_speaker = (const float*)d_in[14];
    const float* emb_turn    = (const float*)d_in[15];
    const float* emb_position= (const float*)d_in[16];
    const float* normal_type = (const float*)d_in[17];
    const float* wr  = (const float*)d_in[18];
    const float* br  = (const float*)d_in[19];
    const float* wz  = (const float*)d_in[20];
    const float* bz  = (const float*)d_in[21];
    const float* wc  = (const float*)d_in[22];
    const float* bc  = (const float*)d_in[23];
    const float* wu  = (const float*)d_in[24];
    const float* bu  = (const float*)d_in[25];
    const float* wq  = (const float*)d_in[26];
    const float* wk  = (const float*)d_in[27];
    const float* wv  = (const float*)d_in[28];
    const float* wo  = (const float*)d_in[29];
    const float* bq  = (const float*)d_in[30];
    const float* bk  = (const float*)d_in[31];
    const float* bvp = (const float*)d_in[32];
    const float* bo  = (const float*)d_in[33];
    const float* wsk = (const float*)d_in[34];
    const float* bsk = (const float*)d_in[35];
    const float* wsv = (const float*)d_in[36];
    const float* bsv = (const float*)d_in[37];
    const float* ln_s = (const float*)d_in[38];
    const float* ln_b = (const float*)d_in[39];
    const float* pln_s = (const float*)d_in[40];
    const float* pln_b = (const float*)d_in[41];
    const float* link_w1 = (const float*)d_in[42];
    const float* link_b1 = (const float*)d_in[43];
    const float* link_w2 = (const float*)d_in[44];
    const float* link_b2 = (const float*)d_in[45];
    const float* lab_w1  = (const float*)d_in[46];
    const float* lab_b1  = (const float*)d_in[47];
    const float* lab_w2  = (const float*)d_in[48];
    const float* lab_b2  = (const float*)d_in[49];

    // ---- carve workspace (all pointers 256B aligned) ----
    size_t off = 0;
    char* base = (char*)d_ws;
    auto carve = [&](size_t bytes) -> void* {
        off = (off + 255) & ~(size_t)255;
        void* p = base + off;
        off += bytes;
        return p;
    };
    float* din      = (float*)carve((size_t)BN * Hh * 4);
    float* xw_f     = (float*)carve((size_t)BN * G3 * 4);
    float* xw_b     = (float*)carve((size_t)BN * G3 * 4);
    float* dout     = (float*)carve((size_t)BN * Hh * 4);
    float* nodes    = (float*)carve((size_t)BN * Hh * 4);
    float* cpath    = (float*)carve((size_t)ROWS * Pp * 4);
    float* sp       = (float*)carve((size_t)ROWS * Pp * 4);
    float* qb       = (float*)carve((size_t)BN * Hh * 4);
    float* kb       = (float*)carve((size_t)BN * Hh * 4);
    float* vb       = (float*)carve((size_t)BN * Hh * 4);
    float* skb      = (float*)carve((size_t)ROWS * Dd * 4);
    float* svb      = (float*)carve((size_t)ROWS * Dd * 4);
    float* actx     = (float*)carve((size_t)BN * Hh * 4);
    float* proj     = (float*)carve((size_t)BN * Hh * 4);
    __bf16* din_bf   = (__bf16*)carve((size_t)BN * Hh * 2);
    __bf16* nodes_bf = (__bf16*)carve((size_t)BN * Hh * 2);
    __bf16* multi_bf = (__bf16*)carve((size_t)BN * CIN_P * 2);
    __bf16* sp_bf    = (__bf16*)carve((size_t)ROWS * Pp * 2);
    __bf16* paths_bf = (__bf16*)carve((size_t)ROWS * Pp * 2);
    __bf16* actx_bf  = (__bf16*)carve((size_t)BN * Hh * 2);
    __bf16* wih_f_bf = (__bf16*)carve((size_t)G3 * Hh * 2);
    __bf16* wih_b_bf = (__bf16*)carve((size_t)G3 * Hh * 2);
    __bf16* wq_bf    = (__bf16*)carve((size_t)Hh * Hh * 2);
    __bf16* wk_bf    = (__bf16*)carve((size_t)Hh * Hh * 2);
    __bf16* wv_bf    = (__bf16*)carve((size_t)Hh * Hh * 2);
    __bf16* wo_bf    = (__bf16*)carve((size_t)Hh * Hh * 2);
    __bf16* wsk_bf   = (__bf16*)carve((size_t)Dd * Pp * 2);
    __bf16* wsv_bf   = (__bf16*)carve((size_t)Dd * Pp * 2);
    __bf16* wr_bf    = (__bf16*)carve((size_t)Pp * CIN_L * 2);
    __bf16* wz_bf    = (__bf16*)carve((size_t)Pp * CIN_L * 2);
    __bf16* wc_bf    = (__bf16*)carve((size_t)Pp * CIN_P * 2);
    __bf16* wu_bf    = (__bf16*)carve((size_t)Pp * Pp * 2);
    __bf16* lw1_bf   = (__bf16*)carve((size_t)Hh * CIN_F * 2);
    __bf16* bw1_bf   = (__bf16*)carve((size_t)Hh * CIN_F * 2);

    auto cvt = [&](const float* s, __bf16* d, int n) {
        cvt_kernel<<<(n + 255) / 256, 256, 0, stream>>>(s, d, n);
    };

    // weight conversions (deterministic, each call)
    cvt(gru_wih_f, wih_f_bf, G3 * Hh);
    cvt(gru_wih_b, wih_b_bf, G3 * Hh);
    cvt(wq, wq_bf, Hh * Hh);
    cvt(wk, wk_bf, Hh * Hh);
    cvt(wv, wv_bf, Hh * Hh);
    cvt(wo, wo_bf, Hh * Hh);
    cvt(wsk, wsk_bf, Dd * Pp);
    cvt(wsv, wsv_bf, Dd * Pp);
    cvt(wr, wr_bf, Pp * CIN_L);
    cvt(wz, wz_bf, Pp * CIN_L);
    cvt(wc, wc_bf, Pp * CIN_P);
    cvt(wu, wu_bf, Pp * Pp);
    cvt(link_w1, lw1_bf, Hh * CIN_F);
    cvt(lab_w1, bw1_bf, Hh * CIN_F);

    // dialog_input = [root; sentences]
    din_kernel<<<(BN * Hh + 255) / 256, 256, 0, stream>>>(sentences, root, din, din_bf);

    // GRU input projections (WMMA): xw = din @ wih^T + bih
    gemm_bf16_kernel<<<dim3(BN / 16, G3 / 64), 32, 0, stream>>>(
        din_bf, Hh, wih_f_bf, Hh, gru_bih_f, xw_f, G3, Hh, 1.0f);
    gemm_bf16_kernel<<<dim3(BN / 16, G3 / 64), 32, 0, stream>>>(
        din_bf, Hh, wih_b_bf, Hh, gru_bih_b, xw_b, G3, Hh, 1.0f);

    // sequential bidirectional GRU
    gru_scan_kernel<<<4, HG, 0, stream>>>(xw_f, xw_b, gru_whh_f, gru_whh_b,
                                          gru_bhh_f, gru_bhh_b, dout);

    nodes_init_kernel<<<(BN * Hh + 255) / 256, 256, 0, stream>>>(dout, nodes, nodes_bf);
    multi_kernel<<<(BN * CIN_P + 255) / 256, 256, 0, stream>>>(din, dout, multi_bf);

    const_sp_kernel<<<ROWS, 128, 0, stream>>>(speakers, turns, graphs, emb_speaker,
                                              emb_turn, emb_position, normal_type,
                                              cpath, sp, sp_bf);

    const float scale = 0.125f;  // D^-0.5
    for (int layer = 0; layer < 3; ++layer) {
        paths_ln_kernel<<<ROWS, 128, 0, stream>>>(sp, sp_bf, cpath, pln_s, pln_b, paths_bf);

        gemm_bf16_kernel<<<dim3(BN / 16, Hh / 64), 32, 0, stream>>>(
            nodes_bf, Hh, wq_bf, Hh, bq, qb, Hh, Hh, scale);
        gemm_bf16_kernel<<<dim3(BN / 16, Hh / 64), 32, 0, stream>>>(
            nodes_bf, Hh, wk_bf, Hh, bk, kb, Hh, Hh, 1.0f);
        gemm_bf16_kernel<<<dim3(BN / 16, Hh / 64), 32, 0, stream>>>(
            nodes_bf, Hh, wv_bf, Hh, bvp, vb, Hh, Hh, 1.0f);

        gemm_bf16_kernel<<<dim3(ROWS / 16, Dd / 64), 32, 0, stream>>>(
            paths_bf, Pp, wsk_bf, Pp, bsk, skb, Dd, Pp, 1.0f);
        gemm_bf16_kernel<<<dim3(ROWS / 16, Dd / 64), 32, 0, stream>>>(
            paths_bf, Pp, wsv_bf, Pp, bsv, svb, Dd, Pp, 1.0f);

        attn_kernel<<<Bq * NHEADS * Nn, 128, 0, stream>>>(qb, kb, vb, skb, svb,
                                                          edu_nums, actx);

        cvt(actx, actx_bf, BN * Hh);
        gemm_bf16_kernel<<<dim3(BN / 16, Hh / 64), 32, 0, stream>>>(
            actx_bf, Hh, wo_bf, Hh, bo, proj, Hh, Hh, 1.0f);

        resid_ln_kernel<<<BN, 256, 0, stream>>>(nodes, nodes_bf, proj, ln_s, ln_b);

        path_update_kernel<<<ROWS / 16, 256, 0, stream>>>(
            nodes_bf, sp_bf, wr_bf, wz_bf, wc_bf, wu_bf, br, bz, bc, bu, sp);
    }

    // refresh sp_bf with final struct_path for the heads
    cvt(sp, sp_bf, ROWS * Pp);

    float* link_out = (float*)d_out;
    float* label_out = link_out + ROWS;
    head_kernel<<<dim3(ROWS / 16, 2), 256, 0, stream>>>(
        multi_bf, sp_bf, lw1_bf, bw1_bf, link_b1, lab_b1,
        link_w2, link_b2, lab_w2, lab_b2, link_out, label_out);
}
